// Gen2SingleInputReadout_50156628082746
// MI455X (gfx1250) — matrix operand order repaired and checked by reading, not compiled, not run on hardware
//
#include <hip/hip_runtime.h>
#include <hip/hip_bf16.h>

// ---------- problem constants ----------
#define T_DIM   2048
#define B_DIM   16
#define IN_DIM  512
#define D_DIM   128
#define CHUNK   64
#define NCHUNK  (T_DIM / CHUNK)   // 32
#define M_TOT   (T_DIM * B_DIM)   // 32768

typedef __attribute__((ext_vector_type(16))) __bf16 v16bf;
typedef __attribute__((ext_vector_type(8)))  __bf16 v8bf;
typedef __attribute__((ext_vector_type(8)))  float  v8f;
typedef __attribute__((__vector_size__(16))) int    v4i;

#define AS_GLOBAL __attribute__((address_space(1)))
#define AS_SHARED __attribute__((address_space(3)))

#if defined(__has_builtin)
#if __has_builtin(__builtin_amdgcn_global_load_async_to_lds_b128)
#define HAVE_ASYNC_LDS 1
#endif
#endif

// ---------- WMMA fragment helpers (CDNA5 16-bit layouts, wave32) ----------
__device__ __forceinline__ int mapCD_M(int lane, int r) {
    return r + ((lane & 16) >> 1);
}
__device__ __forceinline__ v16bf cat8(v8bf lo, v8bf hi) {
    return __builtin_shufflevector(lo, hi, 0,1,2,3,4,5,6,7,8,9,10,11,12,13,14,15);
}
__device__ __forceinline__ v16bf loadA_row(const __bf16* row, int k0, int lane) {
    const int base = k0 + (lane & 16);
    v8bf lo = *(const v8bf*)(row + base);
    v8bf hi = *(const v8bf*)(row + base + 8);
    return cat8(lo, hi);
}
__device__ __forceinline__ v16bf loadB_row(const __bf16* row, int k0, int lane) {
    return *(const v16bf*)(row + k0 + (lane & 16));
}
__device__ __forceinline__ v8f wmma_bf16(v16bf a, v16bf b, v8f c) {
    return __builtin_amdgcn_wmma_f32_16x16x32_bf16(
        false, a, false, b, (short)0, c, false, false);
}
__device__ __forceinline__ float fast_rcp(float x) {
    return __builtin_amdgcn_rcpf(x);
}

// ======================================================================
// Kernel 1: projections  Y = X @ W^T + b  (sigmoid fused for alpha)
// ======================================================================
__global__ __launch_bounds__(256) void proj_kernel(
    const float* __restrict__ x,
    const float* __restrict__ Wv, const float* __restrict__ bv,
    const float* __restrict__ Wk, const float* __restrict__ bk,
    const float* __restrict__ Wq, const float* __restrict__ bq,
    const float* __restrict__ Wa, const float* __restrict__ ba,
    float* __restrict__ vo, float* __restrict__ ko,
    float* __restrict__ qo, float* __restrict__ ao)
{
    __shared__ __align__(32) __bf16 Xs[16][IN_DIM];   // 16 KB

    const int which = blockIdx.y;
    const float* W; const float* bias; float* out; bool sig = false;
    switch (which) {
        case 0:  W = Wv; bias = bv; out = vo; break;
        case 1:  W = Wk; bias = bk; out = ko; break;
        case 2:  W = Wq; bias = bq; out = qo; break;
        default: W = Wa; bias = ba; out = ao; sig = true; break;
    }
    const int tid  = threadIdx.x;
    const int lane = tid & 31;
    const int wave = tid >> 5;
    const int m0   = blockIdx.x * 16;
    const int n0   = wave * 16;
    const int ncol = n0 + (lane & 15);

    for (int e = tid; e < 16 * (IN_DIM / 4); e += 256) {
        const int r = e >> 7, c4 = (e & 127) << 2;
        float4 p = *(const float4*)(x + (size_t)(m0 + r) * IN_DIM + c4);
        Xs[r][c4 + 0] = (__bf16)p.x;
        Xs[r][c4 + 1] = (__bf16)p.y;
        Xs[r][c4 + 2] = (__bf16)p.z;
        Xs[r][c4 + 3] = (__bf16)p.w;
    }
    __syncthreads();

    const float* Wrow = W + (size_t)ncol * IN_DIM;
    v8f acc = {};
    for (int k0 = 0; k0 < IN_DIM; k0 += 32) {
        v16bf a = loadA_row(&Xs[lane & 15][0], k0, lane);
        const int base = k0 + (lane & 16);
        float4 p0 = *(const float4*)(Wrow + base);
        float4 p1 = *(const float4*)(Wrow + base + 4);
        float4 p2 = *(const float4*)(Wrow + base + 8);
        float4 p3 = *(const float4*)(Wrow + base + 12);
        v16bf bb;
        bb[0]=(__bf16)p0.x; bb[1]=(__bf16)p0.y; bb[2]=(__bf16)p0.z; bb[3]=(__bf16)p0.w;
        bb[4]=(__bf16)p1.x; bb[5]=(__bf16)p1.y; bb[6]=(__bf16)p1.z; bb[7]=(__bf16)p1.w;
        bb[8]=(__bf16)p2.x; bb[9]=(__bf16)p2.y; bb[10]=(__bf16)p2.z; bb[11]=(__bf16)p2.w;
        bb[12]=(__bf16)p3.x; bb[13]=(__bf16)p3.y; bb[14]=(__bf16)p3.z; bb[15]=(__bf16)p3.w;
        acc = wmma_bf16(a, bb, acc);
    }
    const float bval = bias[ncol];
    #pragma unroll
    for (int r = 0; r < 8; ++r) {
        float y = acc[r] + bval;
        if (sig) y = fast_rcp(1.0f + __expf(-y));        // v_rcp_f32, no IEEE div
        out[(size_t)(m0 + mapCD_M(lane, r)) * D_DIM + ncol] = y;
    }
}

// ======================================================================
// Kernel 2: gating — cumprod of clipped alpha per (chunk,batch,channel)
// ======================================================================
__global__ __launch_bounds__(128) void gate_kernel(
    const float* __restrict__ kb, const float* __restrict__ qb,
    const float* __restrict__ ab,
    __bf16* __restrict__ Qtg, __bf16* __restrict__ Ktg,
    float* __restrict__ cplg)
{
    const int c = blockIdx.x, b = blockIdx.y, n = threadIdx.x;
    float cp = 1.0f;
    for (int t = 0; t < CHUNK; ++t) {
        const int idx = ((c * CHUNK + t) * B_DIM + b) * D_DIM + n;
        cp *= fmaxf(ab[idx], 1e-8f);
        Qtg[idx] = (__bf16)(qb[idx] * cp);
        Ktg[idx] = (__bf16)(kb[idx] * fast_rcp(cp + 1e-8f));
    }
    cplg[(c * B_DIM + b) * D_DIM + n] = cp;
}

// ======================================================================
// Kernel 3: chunked GLA scan; grid = (8 d-groups, B), block = 128 (4 waves)
// Per chunk:  A = tril(Q~ K~^T);  Y = A V + Q~ S^T;  S = cpl ⊙ (S + V^T K~)
// ======================================================================
__global__ __launch_bounds__(128) void gla_kernel(
    const float*  __restrict__ vb,
    const __bf16* __restrict__ Qtg, const __bf16* __restrict__ Ktg,
    const float*  __restrict__ cplg,
    float* __restrict__ out)
{
    __shared__ __align__(32) __bf16 Qt[CHUNK][D_DIM];    // 16 KB  q~ rows
    __shared__ __align__(32) __bf16 Kt[CHUNK][D_DIM];    // 16 KB  k~ rows
    __shared__ __align__(32) __bf16 VtT[16][CHUNK];      //  2 KB  V^T slice (d-major)
    __shared__ __align__(32) __bf16 Abuf[CHUNK][CHUNK];  //  8 KB  masked A
    __shared__ __align__(32) float  S[16][D_DIM];        //  8 KB  f32 state (master)
    __shared__ __align__(32) __bf16 S16[16][D_DIM];      //  4 KB  bf16 mirror for WMMA
    __shared__ float  cpl[D_DIM];

    const int dg   = blockIdx.x;
    const int b    = blockIdx.y;
    const int d0   = dg * 16;
    const int tid  = threadIdx.x;
    const int lane = tid & 31;
    const int wave = tid >> 5;

    for (int i = tid; i < 16 * D_DIM; i += 128) {
        ((float*)S)[i]    = 0.0f;
        ((__bf16*)S16)[i] = (__bf16)0.0f;
    }
    __syncthreads();

    for (int c = 0; c < NCHUNK; ++c) {
        // ---- phase A: stage chunk into LDS ----
        for (int e = tid; e < CHUNK * 16; e += 128) {            // 16B chunks per row
            const int t = e >> 4, ch = e & 15;
            const size_t rowg = ((size_t)(c * CHUNK + t) * B_DIM + b) * D_DIM;
#ifdef HAVE_ASYNC_LDS
            // global_load_async_to_lds_b128: DMA 16B global -> LDS, no VGPR bounce
            __builtin_amdgcn_global_load_async_to_lds_b128(
                (AS_GLOBAL v4i*)(Qtg + rowg + ch * 8),
                (AS_SHARED v4i*)&Qt[t][ch * 8], 0, 0);
            __builtin_amdgcn_global_load_async_to_lds_b128(
                (AS_GLOBAL v4i*)(Ktg + rowg + ch * 8),
                (AS_SHARED v4i*)&Kt[t][ch * 8], 0, 0);
#else
            ((uint4*)&Qt[t][0])[ch] = ((const uint4*)(Qtg + rowg))[ch];
            ((uint4*)&Kt[t][0])[ch] = ((const uint4*)(Ktg + rowg))[ch];
#endif
        }
        for (int e = tid; e < CHUNK * 16; e += 128) {            // V slice, transposed
            const int t = e >> 4, dn = e & 15;
            VtT[dn][t] = (__bf16) vb[((size_t)(c * CHUNK + t) * B_DIM + b) * D_DIM + d0 + dn];
        }
        if (tid < D_DIM) cpl[tid] = cplg[(c * B_DIM + b) * D_DIM + tid];
#ifdef HAVE_ASYNC_LDS
        asm volatile("s_wait_asynccnt 0x0" ::: "memory");
#endif
        __syncthreads();

        // prefetch next chunk's gated q/k while we compute (global_prefetch_b8)
        if (c + 1 < NCHUNK) {
            const size_t nb = ((size_t)((c + 1) * CHUNK + (tid >> 1)) * B_DIM + b) * D_DIM
                              + (size_t)(tid & 1) * 64;
            __builtin_prefetch(Qtg + nb, 0, 3);
            __builtin_prefetch(Ktg + nb, 0, 3);
        }

        // ---- phase B: A = Q~ K~^T (64x64, K=128); 16 tiles, 4 per wave ----
        for (int tile = wave * 4; tile < wave * 4 + 4; ++tile) {
            const int ti = tile >> 2, tj = tile & 3;
            v8f acc = {};
            #pragma unroll
            for (int k0 = 0; k0 < D_DIM; k0 += 32) {
                v16bf a  = loadA_row(&Qt[ti * 16 + (lane & 15)][0], k0, lane);
                v16bf bb = loadB_row(&Kt[tj * 16 + (lane & 15)][0], k0, lane);
                acc = wmma_bf16(a, bb, acc);
            }
            #pragma unroll
            for (int r = 0; r < 8; ++r) {
                const int t = ti * 16 + mapCD_M(lane, r);
                const int s = tj * 16 + (lane & 15);
                Abuf[t][s] = (__bf16)((s <= t) ? acc[r] : 0.0f);  // causal mask
            }
        }
        __syncthreads();

        // ---- phase C: Y = A V + Q~ S^T  (64x16); 4 t-tiles, 1 per wave ----
        {
            const int ti = wave;
            v8f acc = {};
            #pragma unroll
            for (int k0 = 0; k0 < CHUNK; k0 += 32) {            // intra: A x V
                v16bf a  = loadA_row(&Abuf[ti * 16 + (lane & 15)][0], k0, lane);
                v16bf bb = loadB_row(&VtT[lane & 15][0], k0, lane);
                acc = wmma_bf16(a, bb, acc);
            }
            #pragma unroll
            for (int k0 = 0; k0 < D_DIM; k0 += 32) {            // inter: Q~ x S^T
                v16bf a  = loadA_row(&Qt[ti * 16 + (lane & 15)][0], k0, lane);
                v16bf bb = loadB_row(&S16[lane & 15][0], k0, lane);
                acc = wmma_bf16(a, bb, acc);
            }
            #pragma unroll
            for (int r = 0; r < 8; ++r) {
                const int t = c * CHUNK + ti * 16 + mapCD_M(lane, r);
                out[((size_t)t * B_DIM + b) * D_DIM + d0 + (lane & 15)] = acc[r];
            }
        }
        __syncthreads();  // all reads of S16 done before update

        // ---- phase D: dS = V^T K~ (16x128, K=64); 8 n-tiles, 2 per wave ----
        for (int tile = wave * 2; tile < wave * 2 + 2; ++tile) {
            const int nj = tile;
            v8f acc = {};
            #pragma unroll
            for (int k0 = 0; k0 < CHUNK; k0 += 32) {
                v16bf a = loadA_row(&VtT[lane & 15][0], k0, lane);   // A[m=d][k=s]
                v16bf bf;                                            // B[k=s][n] gather
                #pragma unroll
                for (int i = 0; i < 16; ++i)
                    bf[i] = Kt[k0 + (lane & 16) + i][nj * 16 + (lane & 15)];
                acc = wmma_bf16(a, bf, acc);
            }
            #pragma unroll
            for (int r = 0; r < 8; ++r) {
                const int d = mapCD_M(lane, r);
                const int n = nj * 16 + (lane & 15);
                const float sv = cpl[n] * (S[d][n] + acc[r]);
                S[d][n]   = sv;
                S16[d][n] = (__bf16)sv;
            }
        }
        __syncthreads();  // protect staging buffers before next chunk
    }
}

// ======================================================================
extern "C" void kernel_launch(void* const* d_in, const int* in_sizes, int n_in,
                              void* d_out, int out_size, void* d_ws, size_t ws_size,
                              hipStream_t stream) {
    const float* x  = (const float*)d_in[0];
    const float* Wv = (const float*)d_in[1]; const float* bv = (const float*)d_in[2];
    const float* Wk = (const float*)d_in[3]; const float* bk = (const float*)d_in[4];
    const float* Wq = (const float*)d_in[5]; const float* bq = (const float*)d_in[6];
    const float* Wa = (const float*)d_in[7]; const float* ba = (const float*)d_in[8];
    float* out = (float*)d_out;

    const size_t TBD = (size_t)T_DIM * B_DIM * D_DIM;  // 4.19M elements
    float*  vbuf = (float*)d_ws;
    float*  kbuf = vbuf + TBD;
    float*  qbuf = kbuf + TBD;
    float*  abuf = qbuf + TBD;
    __bf16* Qtg  = (__bf16*)(abuf + TBD);
    __bf16* Ktg  = Qtg + TBD;
    float*  cplg = (float*)(Ktg + TBD);

    dim3 g1(M_TOT / 16, 4);
    proj_kernel<<<g1, 256, 0, stream>>>(x, Wv, bv, Wk, bk, Wq, bq, Wa, ba,
                                        vbuf, kbuf, qbuf, abuf);

    dim3 g2(NCHUNK, B_DIM);
    gate_kernel<<<g2, 128, 0, stream>>>(kbuf, qbuf, abuf, Qtg, Ktg, cplg);

    dim3 g3(D_DIM / 16, B_DIM);
    gla_kernel<<<g3, 128, 0, stream>>>(vbuf, Qtg, Ktg, cplg, out);
}
